// SE3GTLayer_12841952215575
// MI455X (gfx1250) — matrix operand order repaired and checked by reading, not compiled, not run on hardware
//
#include <hip/hip_runtime.h>

typedef __bf16 bf16_t;
typedef __attribute__((ext_vector_type(16))) __bf16 v16bf;
typedef __attribute__((ext_vector_type(8)))  __bf16 v8bf;
typedef __attribute__((ext_vector_type(8)))  float  v8f;

#define NN   8192
#define EE   262144
#define DD   128
#define HH   256
#define DHH  64
#define EDD  16
#define YY   64
#define EINP 288   // 273 padded to 9*32
#define NINP 384

#define V8F_ZERO {0.f,0.f,0.f,0.f,0.f,0.f,0.f,0.f}

#define AS(n) __attribute__((address_space(n)))

// ---- gfx1250 async global->LDS path (guarded; falls back to scalar staging) ----
#if defined(__has_builtin)
#if __has_builtin(__builtin_amdgcn_global_load_async_to_lds_b128) && \
    __has_builtin(__builtin_amdgcn_s_wait_asynccnt)
#define USE_ASYNC_LDS 1
#endif
#endif

#ifdef USE_ASYNC_LDS
// Builtin signature (from hipcc diagnostic): param0 = v4i AS(1)*, param1 = v4i AS(3)*,
// then imm offset + imm cpol.
typedef int v4i_ __attribute__((vector_size(16)));
static __device__ __forceinline__ void async_copy16(const bf16_t* gsrc, bf16_t* ldst) {
  __builtin_amdgcn_global_load_async_to_lds_b128(
      (AS(1) v4i_*)(unsigned long long)(const void*)gsrc,
      (AS(3) v4i_*)(unsigned int)(unsigned long long)(void*)ldst,
      0, 0);
}
#endif

// ---------------- WMMA fragment helpers (CDNA5 wave32 layouts) ----------------
// A 16x32 bf16: lane l -> row l&15; halves 0..7 = K [(l>>4)*8 .. +7], halves 8..15 = K [16+(l>>4)*8 .. +7]
static __device__ __forceinline__ v16bf load_a16x32(const bf16_t* tile, int ld) {
  const int l = threadIdx.x & 31;
  const bf16_t* r = tile + (l & 15) * ld + ((l >> 4) & 1) * 8;
  v8bf lo = *(const v8bf*)(r);
  v8bf hi = *(const v8bf*)(r + 16);
  v16bf a;
#pragma unroll
  for (int i = 0; i < 8; ++i) { a[i] = lo[i]; a[i + 8] = hi[i]; }
  return a;
}

// B 32x16 bf16 loaded from B^T stored row-major [N,K]: lane l -> col l&15; halves 0..7 = K [(l>>4)*8 .. +7], halves 8..15 = K [16+(l>>4)*8 .. +7]
static __device__ __forceinline__ v16bf load_b32x16(const bf16_t* tileT, int ld) {
  const int l = threadIdx.x & 31;
  const bf16_t* r = tileT + (l & 15) * ld + ((l >> 4) & 1) * 8;
  v8bf lo = *(const v8bf*)(r);
  v8bf hi = *(const v8bf*)(r + 16);
  v16bf b;
#pragma unroll
  for (int i = 0; i < 8; ++i) { b[i] = lo[i]; b[i + 8] = hi[i]; }
  return b;
}

static __device__ __forceinline__ v8f wmma_bf16(v16bf a, v16bf b, v8f c) {
  return __builtin_amdgcn_wmma_f32_16x16x32_bf16(false, a, false, b, (short)0, c, false, false);
}

static __device__ __forceinline__ float siluf(float v) {
  return v / (1.0f + __expf(-v));
}

// ---------------- prep kernels ----------------
__global__ void k_convert_bf(const float* __restrict__ s, bf16_t* __restrict__ d, int n) {
  int i = blockIdx.x * blockDim.x + threadIdx.x;
  if (i < n) d[i] = (bf16_t)s[i];
}

// w [fin,fout] f32 -> wt [fout,finPad] bf16 (transposed, K-padded with zeros)
__global__ void k_convert_wT(const float* __restrict__ w, bf16_t* __restrict__ wt,
                             int fin, int fout, int finPad) {
  int idx = blockIdx.x * blockDim.x + threadIdx.x;
  int total = fout * finPad;
  if (idx >= total) return;
  int o = idx / finPad, i = idx - o * finPad;
  wt[idx] = (i < fin) ? (bf16_t)w[i * fout + o] : (bf16_t)0.0f;
}

// ---------------- edge MLP + fused segment_sum (dominant GEMM) ----------------
__launch_bounds__(128)
__global__ void k_edge(const bf16_t* __restrict__ xbf, const float* __restrict__ coords,
                       const float* __restrict__ edge_attr, const int* __restrict__ eidx,
                       const bf16_t* __restrict__ ew1T, const float* __restrict__ eb1,
                       const bf16_t* __restrict__ ew2T, const float* __restrict__ eb2,
                       float* __restrict__ agg) {
  __shared__ __align__(16) bf16_t sM[32][EINP];  // 18 KB
  __shared__ __align__(16) bf16_t sH[32][HH];    // 16 KB
  const int tid = threadIdx.x;
  const int l = tid & 31;
  const int wave = tid >> 5;
  const int e0 = blockIdx.x * 32;
  const int khalf = (l >> 4) & 1;

  __builtin_prefetch(edge_attr + (size_t)(e0 + 32) * EDD, 0, 1);  // global_prefetch_b8

#ifdef USE_ASYNC_LDS
  // x_i / x_j gather: async global->LDS, 16B per op (raw bf16, no conversion needed)
  for (int i = tid; i < 32 * 32; i += 128) {
    int e = i >> 5, c = i & 31;                 // c<16: x_row chunk, else x_col chunk
    int eg = e0 + e;
    int node = (c < 16) ? eidx[eg] : eidx[EE + eg];
    int f = (c & 15) * 8;                       // halves within the 128-wide segment
    async_copy16(xbf + (size_t)node * DD + f, &sM[e][(c < 16 ? 0 : DD) + f]);
  }
  // radial + edge_attr + pad (needs f32 -> bf16 conversion): scalar path
  for (int i = tid; i < 32 * 32; i += 128) {
    int e = i >> 5, f = 2 * DD + (i & 31);
    int eg = e0 + e;
    float v;
    if (f == 2 * DD) {
      int rn = eidx[eg], cn = eidx[EE + eg];
      float dx = coords[rn * 3 + 0] - coords[cn * 3 + 0];
      float dy = coords[rn * 3 + 1] - coords[cn * 3 + 1];
      float dz = coords[rn * 3 + 2] - coords[cn * 3 + 2];
      v = dx * dx + dy * dy + dz * dz;
    } else if (f < 273) {
      v = edge_attr[(size_t)eg * EDD + (f - 257)];
    } else {
      v = 0.f;
    }
    sM[e][f] = (bf16_t)v;
  }
  __builtin_amdgcn_s_wait_asynccnt(0);
  __syncthreads();
#else
  // stage m_in = [x_row | x_col | radial | edge_attr | pad]
  for (int i = tid; i < 32 * EINP; i += 128) {
    int e = i / EINP, f = i - e * EINP;
    int eg = e0 + e;
    int rn = eidx[eg];
    int cn = eidx[EE + eg];
    float v;
    if (f < DD)            v = (float)xbf[rn * DD + f];
    else if (f < 2 * DD)   v = (float)xbf[cn * DD + (f - DD)];
    else if (f == 2 * DD) {
      float dx = coords[rn * 3 + 0] - coords[cn * 3 + 0];
      float dy = coords[rn * 3 + 1] - coords[cn * 3 + 1];
      float dz = coords[rn * 3 + 2] - coords[cn * 3 + 2];
      v = dx * dx + dy * dy + dz * dz;
    } else if (f < 273)    v = edge_attr[(size_t)eg * EDD + (f - 257)];
    else                   v = 0.f;
    sM[e][f] = (bf16_t)v;
  }
  __syncthreads();
#endif

  const int mtile = wave & 1;   // 2 row-tiles of 16 edges
  const int nh = wave >> 1;     // split 16 N-tiles across 2 waves

  // layer 1: [32,288] @ [288,256]
  v16bf A[9];
#pragma unroll
  for (int kt = 0; kt < 9; ++kt) A[kt] = load_a16x32(&sM[mtile * 16][kt * 32], EINP);
  for (int nt = 0; nt < 8; ++nt) {
    int ntile = nh * 8 + nt;
    v8f acc = V8F_ZERO;
#pragma unroll
    for (int kt = 0; kt < 9; ++kt)
      acc = wmma_bf16(A[kt], load_b32x16(ew1T + (ntile * 16) * EINP + kt * 32, EINP), acc);
    float b = eb1[ntile * 16 + (l & 15)];
#pragma unroll
    for (int r = 0; r < 8; ++r)
      sH[mtile * 16 + khalf * 8 + r][ntile * 16 + (l & 15)] = (bf16_t)siluf(acc[r] + b);
  }
  __syncthreads();

  // layer 2: [32,256] @ [256,256], then silu + scatter-add into agg[row]
  v16bf A2[8];
#pragma unroll
  for (int kt = 0; kt < 8; ++kt) A2[kt] = load_a16x32(&sH[mtile * 16][kt * 32], HH);
  int rnode[8];
#pragma unroll
  for (int r = 0; r < 8; ++r) rnode[r] = eidx[e0 + mtile * 16 + khalf * 8 + r];
  for (int nt = 0; nt < 8; ++nt) {
    int ntile = nh * 8 + nt;
    v8f acc = V8F_ZERO;
#pragma unroll
    for (int kt = 0; kt < 8; ++kt)
      acc = wmma_bf16(A2[kt], load_b32x16(ew2T + (ntile * 16) * HH + kt * 32, HH), acc);
    float b = eb2[ntile * 16 + (l & 15)];
#pragma unroll
    for (int r = 0; r < 8; ++r) {
      float v = siluf(acc[r] + b);
      (void)__hip_atomic_fetch_add(&agg[(size_t)rnode[r] * HH + ntile * 16 + (l & 15)], v,
                                   __ATOMIC_RELAXED, __HIP_MEMORY_SCOPE_AGENT);
    }
  }
}

// ---------------- node MLP (residual) ----------------
__launch_bounds__(128)
__global__ void k_node(const bf16_t* __restrict__ xbf, const float* __restrict__ x,
                       const float* __restrict__ agg,
                       const bf16_t* __restrict__ nw1T, const float* __restrict__ nb1,
                       const bf16_t* __restrict__ nw2T, const float* __restrict__ nb2,
                       float* __restrict__ xgnn) {
  __shared__ __align__(16) bf16_t sM[32][NINP];  // 24 KB
  __shared__ __align__(16) bf16_t sH[32][HH];    // 16 KB
  const int tid = threadIdx.x;
  const int l = tid & 31;
  const int wave = tid >> 5;
  const int n0b = blockIdx.x * 32;
  const int khalf = (l >> 4) & 1;

#ifdef USE_ASYNC_LDS
  // x part: contiguous bf16, async global->LDS
  for (int i = tid; i < 32 * 16; i += 128) {
    int nloc = i >> 4, c = i & 15;
    async_copy16(xbf + (size_t)(n0b + nloc) * DD + c * 8, &sM[nloc][c * 8]);
  }
  // agg part: f32 -> bf16 conversion, scalar
  for (int i = tid; i < 32 * HH; i += 128) {
    int nloc = i >> 8, f = i & (HH - 1);
    sM[nloc][DD + f] = (bf16_t)agg[(size_t)(n0b + nloc) * HH + f];
  }
  __builtin_amdgcn_s_wait_asynccnt(0);
  __syncthreads();
#else
  for (int i = tid; i < 32 * NINP; i += 128) {
    int nloc = i / NINP, f = i - nloc * NINP;
    int node = n0b + nloc;
    float v = (f < DD) ? (float)xbf[node * DD + f] : agg[(size_t)node * HH + (f - DD)];
    sM[nloc][f] = (bf16_t)v;
  }
  __syncthreads();
#endif

  const int mtile = wave & 1;
  const int nh = wave >> 1;

  v16bf A[12];
#pragma unroll
  for (int kt = 0; kt < 12; ++kt) A[kt] = load_a16x32(&sM[mtile * 16][kt * 32], NINP);
  for (int nt = 0; nt < 8; ++nt) {
    int ntile = nh * 8 + nt;
    v8f acc = V8F_ZERO;
#pragma unroll
    for (int kt = 0; kt < 12; ++kt)
      acc = wmma_bf16(A[kt], load_b32x16(nw1T + (ntile * 16) * NINP + kt * 32, NINP), acc);
    float b = nb1[ntile * 16 + (l & 15)];
#pragma unroll
    for (int r = 0; r < 8; ++r)
      sH[mtile * 16 + khalf * 8 + r][ntile * 16 + (l & 15)] = (bf16_t)siluf(acc[r] + b);
  }
  __syncthreads();

  v16bf A2[8];
#pragma unroll
  for (int kt = 0; kt < 8; ++kt) A2[kt] = load_a16x32(&sH[mtile * 16][kt * 32], HH);
  for (int nt = 0; nt < 4; ++nt) {
    int ntile = nh * 4 + nt;
    v8f acc = V8F_ZERO;
#pragma unroll
    for (int kt = 0; kt < 8; ++kt)
      acc = wmma_bf16(A2[kt], load_b32x16(nw2T + (ntile * 16) * HH + kt * 32, HH), acc);
    int col = ntile * 16 + (l & 15);
    float b = nb2[col];
#pragma unroll
    for (int r = 0; r < 8; ++r) {
      int node = n0b + mtile * 16 + khalf * 8 + r;
      xgnn[(size_t)node * DD + col] = x[(size_t)node * DD + col] + acc[r] + b;
    }
  }
}

// ---------------- q/k/v projections ----------------
__launch_bounds__(128)
__global__ void k_qkv(const bf16_t* __restrict__ xbf,
                      const bf16_t* __restrict__ wqT, const bf16_t* __restrict__ wkT,
                      const bf16_t* __restrict__ wvT,
                      bf16_t* __restrict__ qbf, bf16_t* __restrict__ kbf,
                      bf16_t* __restrict__ vT) {
  const int l = threadIdx.x & 31;
  const int wave = threadIdx.x >> 5;
  const int node0 = blockIdx.x * 64 + wave * 16;
  const int khalf = (l >> 4) & 1;
  v16bf A[4];
#pragma unroll
  for (int kt = 0; kt < 4; ++kt) A[kt] = load_a16x32(xbf + (size_t)node0 * DD + kt * 32, DD);
  const bf16_t* Ws[3] = {wqT, wkT, wvT};
#pragma unroll
  for (int m = 0; m < 3; ++m) {
    for (int nt = 0; nt < 4; ++nt) {
      v8f acc = V8F_ZERO;
#pragma unroll
      for (int kt = 0; kt < 4; ++kt)
        acc = wmma_bf16(A[kt], load_b32x16(Ws[m] + (nt * 16) * DD + kt * 32, DD), acc);
      int col = nt * 16 + (l & 15);
#pragma unroll
      for (int r = 0; r < 8; ++r) {
        int row = node0 + khalf * 8 + r;
        bf16_t v = (bf16_t)acc[r];
        if (m == 0)      qbf[(size_t)row * DHH + col] = v;
        else if (m == 1) kbf[(size_t)row * DHH + col] = v;
        else             vT[(size_t)col * NN + row] = v;   // transposed for PV B-fragments
      }
    }
  }
}

// ---------------- radius-masked flash attention (one 16-row tile per wave) ----------------
__launch_bounds__(128)
__global__ void k_attn(const bf16_t* __restrict__ qbf, const bf16_t* __restrict__ kbf,
                       const bf16_t* __restrict__ vT, const float* __restrict__ coords,
                       bf16_t* __restrict__ obf) {
  __shared__ __align__(16) bf16_t sP[4][16][32];  // per-wave P tile
  const int l = threadIdx.x & 31;
  const int wave = threadIdx.x >> 5;
  const int m0 = (blockIdx.x * 4 + wave) * 16;
  const int khalf = (l >> 4) & 1;

  v16bf Aq0 = load_a16x32(qbf + (size_t)m0 * DHH, DHH);
  v16bf Aq1 = load_a16x32(qbf + (size_t)m0 * DHH + 32, DHH);

  float qcx[8], qcy[8], qcz[8];
#pragma unroll
  for (int r = 0; r < 8; ++r) {
    int qr = m0 + khalf * 8 + r;
    qcx[r] = coords[qr * 3 + 0];
    qcy[r] = coords[qr * 3 + 1];
    qcz[r] = coords[qr * 3 + 2];
  }
  v8f acc0 = V8F_ZERO, acc1 = V8F_ZERO, acc2 = V8F_ZERO, acc3 = V8F_ZERO;
  float mstat[8], lstat[8];
#pragma unroll
  for (int r = 0; r < 8; ++r) { mstat[r] = -3.0e38f; lstat[r] = 0.f; }

  for (int n0 = 0; n0 < NN; n0 += 32) {
    __builtin_prefetch(kbf + (size_t)(n0 + 32) * DHH, 0, 1);
    v8f S0 = V8F_ZERO, S1 = V8F_ZERO;
    S0 = wmma_bf16(Aq0, load_b32x16(kbf + (size_t)n0 * DHH, DHH), S0);
    S0 = wmma_bf16(Aq1, load_b32x16(kbf + (size_t)n0 * DHH + 32, DHH), S0);
    S1 = wmma_bf16(Aq0, load_b32x16(kbf + (size_t)(n0 + 16) * DHH, DHH), S1);
    S1 = wmma_bf16(Aq1, load_b32x16(kbf + (size_t)(n0 + 16) * DHH + 32, DHH), S1);

    int kc0 = n0 + (l & 15), kc1 = kc0 + 16;
    float k0x = coords[kc0 * 3], k0y = coords[kc0 * 3 + 1], k0z = coords[kc0 * 3 + 2];
    float k1x = coords[kc1 * 3], k1y = coords[kc1 * 3 + 1], k1z = coords[kc1 * 3 + 2];

    float s0[8], s1[8];
#pragma unroll
    for (int r = 0; r < 8; ++r) {
      float a = S0[r] * 0.125f;  // DH^-0.5
      float dx = qcx[r] - k0x, dy = qcy[r] - k0y, dz = qcz[r] - k0z;
      if (dx * dx + dy * dy + dz * dz > 100.f) a = -1.0e9f;
      float b = S1[r] * 0.125f;
      dx = qcx[r] - k1x; dy = qcy[r] - k1y; dz = qcz[r] - k1z;
      if (dx * dx + dy * dy + dz * dz > 100.f) b = -1.0e9f;
      s0[r] = a; s1[r] = b;
    }
#pragma unroll
    for (int r = 0; r < 8; ++r) {
      float cm = fmaxf(s0[r], s1[r]);
#pragma unroll
      for (int o = 1; o < 16; o <<= 1) cm = fmaxf(cm, __shfl_xor(cm, o, 32));
      float mnew = fmaxf(mstat[r], cm);
      float alpha = __expf(mstat[r] - mnew);
      float p0 = __expf(s0[r] - mnew);
      float p1 = __expf(s1[r] - mnew);
      float ps = p0 + p1;
#pragma unroll
      for (int o = 1; o < 16; o <<= 1) ps += __shfl_xor(ps, o, 32);
      lstat[r] = lstat[r] * alpha + ps;
      mstat[r] = mnew;
      acc0[r] *= alpha; acc1[r] *= alpha; acc2[r] *= alpha; acc3[r] *= alpha;
      int rit = khalf * 8 + r;
      sP[wave][rit][l & 15] = (bf16_t)p0;
      sP[wave][rit][16 + (l & 15)] = (bf16_t)p1;
    }
    v16bf Ap = load_a16x32(&sP[wave][0][0], 32);
    acc0 = wmma_bf16(Ap, load_b32x16(vT + (size_t)(0 * 16) * NN + n0, NN), acc0);
    acc1 = wmma_bf16(Ap, load_b32x16(vT + (size_t)(1 * 16) * NN + n0, NN), acc1);
    acc2 = wmma_bf16(Ap, load_b32x16(vT + (size_t)(2 * 16) * NN + n0, NN), acc2);
    acc3 = wmma_bf16(Ap, load_b32x16(vT + (size_t)(3 * 16) * NN + n0, NN), acc3);
  }
#pragma unroll
  for (int r = 0; r < 8; ++r) {
    float inv = 1.0f / lstat[r];
    int row = m0 + khalf * 8 + r;
    obf[(size_t)row * DHH +  0 + (l & 15)] = (bf16_t)(acc0[r] * inv);
    obf[(size_t)row * DHH + 16 + (l & 15)] = (bf16_t)(acc1[r] * inv);
    obf[(size_t)row * DHH + 32 + (l & 15)] = (bf16_t)(acc2[r] * inv);
    obf[(size_t)row * DHH + 48 + (l & 15)] = (bf16_t)(acc3[r] * inv);
  }
}

// ---------------- combiner: o@wo + xgnn -> relu(Linear) -> LayerNorm -> Linear ----------------
__launch_bounds__(128)
__global__ void k_comb(const bf16_t* __restrict__ obf, const bf16_t* __restrict__ woT,
                       const float* __restrict__ xgnn,
                       const bf16_t* __restrict__ cmw1T, const float* __restrict__ cmb1,
                       const float* __restrict__ ln_g, const float* __restrict__ ln_b,
                       const bf16_t* __restrict__ cmw2T, const float* __restrict__ cmb2,
                       float* __restrict__ out) {
  __shared__ __align__(16) bf16_t sC[32][DD];   // 8 KB
  __shared__ __align__(16) float  sHf[32][HH];  // 32 KB
  __shared__ __align__(16) bf16_t sN[32][HH];   // 16 KB
  __shared__ float sMu[32], sRs[32];
  const int tid = threadIdx.x;
  const int l = tid & 31;
  const int wave = tid >> 5;
  const int n0b = blockIdx.x * 32;
  const int mtile = wave & 1;
  const int nh = wave >> 1;
  const int khalf = (l >> 4) & 1;

  // combined = xgnn + o @ wo
  v16bf Ao[2];
#pragma unroll
  for (int kt = 0; kt < 2; ++kt)
    Ao[kt] = load_a16x32(obf + (size_t)(n0b + mtile * 16) * DHH + kt * 32, DHH);
  for (int nt = 0; nt < 4; ++nt) {
    int ntile = nh * 4 + nt;
    v8f acc = V8F_ZERO;
#pragma unroll
    for (int kt = 0; kt < 2; ++kt)
      acc = wmma_bf16(Ao[kt], load_b32x16(woT + (ntile * 16) * DHH + kt * 32, DHH), acc);
    int col = ntile * 16 + (l & 15);
#pragma unroll
    for (int r = 0; r < 8; ++r) {
      int rl = mtile * 16 + khalf * 8 + r;
      sC[rl][col] = (bf16_t)(acc[r] + xgnn[(size_t)(n0b + rl) * DD + col]);
    }
  }
  __syncthreads();

  // h2 = relu(combined @ cmw1 + cmb1)
  v16bf Ac[4];
#pragma unroll
  for (int kt = 0; kt < 4; ++kt) Ac[kt] = load_a16x32(&sC[mtile * 16][kt * 32], DD);
  for (int nt = 0; nt < 8; ++nt) {
    int ntile = nh * 8 + nt;
    v8f acc = V8F_ZERO;
#pragma unroll
    for (int kt = 0; kt < 4; ++kt)
      acc = wmma_bf16(Ac[kt], load_b32x16(cmw1T + (ntile * 16) * DD + kt * 32, DD), acc);
    int col = ntile * 16 + (l & 15);
    float b = cmb1[col];
#pragma unroll
    for (int r = 0; r < 8; ++r)
      sHf[mtile * 16 + khalf * 8 + r][col] = fmaxf(acc[r] + b, 0.f);
  }
  __syncthreads();

  // LayerNorm statistics (population variance, matches jnp.var)
  if (tid < 32) {
    float s = 0.f, ss = 0.f;
    for (int c = 0; c < HH; ++c) { float v = sHf[tid][c]; s += v; ss += v * v; }
    float mu = s * (1.0f / HH);
    float var = ss * (1.0f / HH) - mu * mu;
    sMu[tid] = mu;
    sRs[tid] = rsqrtf(var + 1e-5f);
  }
  __syncthreads();
  for (int i = tid; i < 32 * HH; i += 128) {
    int rl = i >> 8, c = i & (HH - 1);
    sN[rl][c] = (bf16_t)((sHf[rl][c] - sMu[rl]) * sRs[rl] * ln_g[c] + ln_b[c]);
  }
  __syncthreads();

  // out = h2n @ cmw2 + cmb2
  v16bf An[8];
#pragma unroll
  for (int kt = 0; kt < 8; ++kt) An[kt] = load_a16x32(&sN[mtile * 16][kt * 32], HH);
  for (int nt = 0; nt < 2; ++nt) {
    int ntile = nh * 2 + nt;
    v8f acc = V8F_ZERO;
#pragma unroll
    for (int kt = 0; kt < 8; ++kt)
      acc = wmma_bf16(An[kt], load_b32x16(cmw2T + (ntile * 16) * HH + kt * 32, HH), acc);
    int col = ntile * 16 + (l & 15);
    float b = cmb2[col];
#pragma unroll
    for (int r = 0; r < 8; ++r) {
      int node = n0b + mtile * 16 + khalf * 8 + r;
      out[(size_t)node * YY + col] = acc[r] + b;
    }
  }
}

// ---------------- host launch ----------------
extern "C" void kernel_launch(void* const* d_in, const int* in_sizes, int n_in,
                              void* d_out, int out_size, void* d_ws, size_t ws_size,
                              hipStream_t stream) {
  const float* x         = (const float*)d_in[0];
  const float* coords    = (const float*)d_in[1];
  const float* edge_attr = (const float*)d_in[2];
  const int*   eidx      = (const int*)d_in[3];
  const float* ew1  = (const float*)d_in[4];
  const float* eb1  = (const float*)d_in[5];
  const float* ew2  = (const float*)d_in[6];
  const float* eb2  = (const float*)d_in[7];
  const float* nw1  = (const float*)d_in[8];
  const float* nb1  = (const float*)d_in[9];
  const float* nw2  = (const float*)d_in[10];
  const float* nb2  = (const float*)d_in[11];
  const float* wq   = (const float*)d_in[12];
  const float* wk   = (const float*)d_in[13];
  const float* wv   = (const float*)d_in[14];
  const float* wo   = (const float*)d_in[15];
  const float* cmw1 = (const float*)d_in[16];
  const float* cmb1 = (const float*)d_in[17];
  const float* ln_g = (const float*)d_in[18];
  const float* ln_b = (const float*)d_in[19];
  const float* cmw2 = (const float*)d_in[20];
  const float* cmb2 = (const float*)d_in[21];

  unsigned char* wsp = (unsigned char*)d_ws;
  auto carve = [&](size_t bytes) -> void* {
    void* p = (void*)wsp;
    wsp += (bytes + 255) & ~(size_t)255;
    return p;
  };
  bf16_t* xbf   = (bf16_t*)carve((size_t)NN * DD * 2);
  bf16_t* ew1T  = (bf16_t*)carve((size_t)HH * EINP * 2);
  bf16_t* ew2T  = (bf16_t*)carve((size_t)HH * HH * 2);
  bf16_t* nw1T  = (bf16_t*)carve((size_t)HH * NINP * 2);
  bf16_t* nw2T  = (bf16_t*)carve((size_t)DD * HH * 2);
  bf16_t* wqT   = (bf16_t*)carve((size_t)DHH * DD * 2);
  bf16_t* wkT   = (bf16_t*)carve((size_t)DHH * DD * 2);
  bf16_t* wvT   = (bf16_t*)carve((size_t)DHH * DD * 2);
  bf16_t* woT   = (bf16_t*)carve((size_t)DD * DHH * 2);
  bf16_t* cmw1T = (bf16_t*)carve((size_t)HH * DD * 2);
  bf16_t* cmw2T = (bf16_t*)carve((size_t)YY * HH * 2);
  float*  agg   = (float*)carve((size_t)NN * HH * 4);
  float*  xgnn  = (float*)carve((size_t)NN * DD * 4);
  bf16_t* qbf   = (bf16_t*)carve((size_t)NN * DHH * 2);
  bf16_t* kbf   = (bf16_t*)carve((size_t)NN * DHH * 2);
  bf16_t* vT    = (bf16_t*)carve((size_t)DHH * NN * 2);
  bf16_t* obf   = (bf16_t*)carve((size_t)NN * DHH * 2);

  (void)hipMemsetAsync(agg, 0, (size_t)NN * HH * 4, stream);

  { int n = NN * DD; k_convert_bf<<<(n + 255) / 256, 256, 0, stream>>>(x, xbf, n); }
  auto cw = [&](const float* w, bf16_t* wt, int fin, int fout, int finPad) {
    int total = fout * finPad;
    k_convert_wT<<<(total + 255) / 256, 256, 0, stream>>>(w, wt, fin, fout, finPad);
  };
  cw(ew1, ew1T, 273, HH, EINP);
  cw(ew2, ew2T, HH, HH, HH);
  cw(nw1, nw1T, NINP, HH, NINP);
  cw(nw2, nw2T, HH, DD, HH);
  cw(wq, wqT, DD, DHH, DD);
  cw(wk, wkT, DD, DHH, DD);
  cw(wv, wvT, DD, DHH, DD);
  cw(wo, woT, DHH, DD, DHH);
  cw(cmw1, cmw1T, DD, HH, DD);
  cw(cmw2, cmw2T, HH, YY, HH);

  k_edge<<<EE / 32, 128, 0, stream>>>(xbf, coords, edge_attr, eidx, ew1T, eb1, ew2T, eb2, agg);
  k_node<<<NN / 32, 128, 0, stream>>>(xbf, x, agg, nw1T, nb1, nw2T, nb2, xgnn);
  k_qkv<<<NN / 64, 128, 0, stream>>>(xbf, wqT, wkT, wvT, qbf, kbf, vT);
  k_attn<<<NN / 64, 128, 0, stream>>>(qbf, kbf, vT, coords, obf);
  k_comb<<<NN / 32, 128, 0, stream>>>(obf, woT, xgnn, cmw1T, cmb1, ln_g, ln_b, cmw2T, cmb2,
                                      (float*)d_out);
  (void)in_sizes; (void)n_in; (void)out_size; (void)ws_size;
}
